// RPGNN_58205396795406
// MI455X (gfx1250) — matrix operand order repaired and checked by reading, not compiled, not run on hardware
//
#include <hip/hip_runtime.h>
#include <hip/hip_bf16.h>
#include <stdint.h>

// ---------------- problem constants (from reference) ----------------
#define NN      100000      // nodes
#define EE      600000      // edges
#define DD      128         // input feature dim
#define HH      256         // hidden dim
#define OO      256         // output dim
#define GG      2000        // graphs
#define K1      138         // D + FIXED_SIZE
#define K1P     160         // K1 padded to multiple of 32 for wmma K-chunks
#define MPAD    100096      // NN padded to multiple of 64 (block M-tile)
#define HALF    50000       // NN/2 for threefry counter split
#define FIXEDSZ 10

typedef __attribute__((ext_vector_type(16))) __bf16 v16bf;
typedef __attribute__((ext_vector_type(8)))  float  v8f;

union Frag { uint4 u[2]; v16bf v; };

// ---------------- helpers ----------------
__device__ __forceinline__ unsigned short f2bf(float f) {
    // round-to-nearest-even f32 -> bf16 (finite inputs)
    uint32_t u = __float_as_uint(f);
    u += 0x7FFFu + ((u >> 16) & 1u);
    return (unsigned short)(u >> 16);
}

// Threefry-2x32, 20 rounds (JAX-compatible key schedule)
__device__ __forceinline__ void threefry2x32(uint32_t k0, uint32_t k1,
                                             uint32_t c0, uint32_t c1,
                                             uint32_t& o0, uint32_t& o1) {
    const uint32_t ks2 = k0 ^ k1 ^ 0x1BD11BDAu;
    uint32_t x0 = c0 + k0, x1 = c1 + k1;
    const int R[8] = {13, 15, 26, 6, 17, 29, 16, 24};
#define TF_ROUND(r) { x0 += x1; x1 = (x1 << (r)) | (x1 >> (32 - (r))); x1 ^= x0; }
    TF_ROUND(R[0]) TF_ROUND(R[1]) TF_ROUND(R[2]) TF_ROUND(R[3]) x0 += k1;  x1 += ks2 + 1u;
    TF_ROUND(R[4]) TF_ROUND(R[5]) TF_ROUND(R[6]) TF_ROUND(R[7]) x0 += ks2; x1 += k0 + 2u;
    TF_ROUND(R[0]) TF_ROUND(R[1]) TF_ROUND(R[2]) TF_ROUND(R[3]) x0 += k0;  x1 += k1 + 3u;
    TF_ROUND(R[4]) TF_ROUND(R[5]) TF_ROUND(R[6]) TF_ROUND(R[7]) x0 += k1;  x1 += ks2 + 4u;
    TF_ROUND(R[0]) TF_ROUND(R[1]) TF_ROUND(R[2]) TF_ROUND(R[3]) x0 += ks2; x1 += k0 + 5u;
#undef TF_ROUND
    o0 = x0; o1 = x1;
}

// ---------------- kernels ----------------
__global__ void k_zero(float* __restrict__ p, long n) {
    long i = (long)blockIdx.x * blockDim.x + threadIdx.x;
    long stride = (long)gridDim.x * blockDim.x;
    for (; i < n; i += stride) p[i] = 0.0f;
}

// starts[g] = lower_bound(batch_ids, g); starts[GG] = NN. batch_ids is sorted.
__global__ void k_starts(const int* __restrict__ batch, int* __restrict__ starts) {
    int g = blockIdx.x * blockDim.x + threadIdx.x;
    if (g > GG) return;
    if (g == GG) { starts[GG] = NN; return; }
    int lo = 0, hi = NN;
    while (lo < hi) {
        int mid = (lo + hi) >> 1;
        if (batch[mid] < g) lo = mid + 1; else hi = mid;
    }
    starts[g] = lo;
}

// r[i] = jax.random.uniform(fold_in(key(42), 0), (N,))[i]
__global__ void k_rng(float* __restrict__ r) {
    int i = blockIdx.x * blockDim.x + threadIdx.x;
    if (i >= NN) return;
    uint32_t k0, k1;
    threefry2x32(0u, 42u, 0u, 0u, k0, k1);          // fold_in(key(42), 0)
    bool lohalf = (i < HALF);
    uint32_t c0 = lohalf ? (uint32_t)i : (uint32_t)(i - HALF);
    uint32_t c1 = lohalf ? (uint32_t)(i + HALF) : (uint32_t)i;
    uint32_t o0, o1;
    threefry2x32(k0, k1, c0, c1, o0, o1);
    uint32_t bits = lohalf ? o0 : o1;
    r[i] = __uint_as_float((bits >> 9) | 0x3F800000u) - 1.0f;
}

// rank within graph via in-segment comparison against f32 key (g + r), stable tie-break
__global__ void k_rank(const int* __restrict__ batch, const int* __restrict__ starts,
                       const float* __restrict__ r, int* __restrict__ rmod) {
    int i = blockIdx.x * blockDim.x + threadIdx.x;
    if (i >= NN) return;
    int g = batch[i];
    int s = starts[g], e = starts[g + 1];
    float gf = (float)g;
    float ki = gf + r[i];
    int cnt = 0;
    for (int j = s; j < e; ++j) {
        float kj = gf + r[j];
        cnt += (kj < ki) || (kj == ki && j < i);
    }
    rmod[i] = cnt % FIXEDSZ;
}

// layer-1 messages: mx[dst][0:128] += x[src]; mx[dst][128 + rmod[src]] += 1
// 32 lanes per edge, 4 contiguous floats per lane (float4 gather; L2-resident x)
__global__ void k_edge1(const int* __restrict__ src, const int* __restrict__ dst,
                        const float* __restrict__ x, const int* __restrict__ rmod,
                        float* __restrict__ mx) {
    int t = blockIdx.x * blockDim.x + threadIdx.x;
    int e = t >> 5, lane = t & 31;
    if (e >= EE) return;
    int s = src[e], d = dst[e];
    const float4 v = *(const float4*)(x + (size_t)s * DD + lane * 4);
    float* mrow = mx + (size_t)d * K1 + lane * 4;
    atomicAdd(mrow + 0, v.x);
    atomicAdd(mrow + 1, v.y);
    atomicAdd(mrow + 2, v.z);
    atomicAdd(mrow + 3, v.w);
    if (lane == 0) atomicAdd(mx + (size_t)d * K1 + DD + rmod[s], 1.0f);
}

// A1[v] = bf16([x|onehot](v) + mx[v]), zero-padded to K1P; pad rows zeroed
__global__ void k_buildA1(const float* __restrict__ x, const float* __restrict__ mx,
                          const int* __restrict__ rmod, unsigned short* __restrict__ A1) {
    int v = blockIdx.x * blockDim.x + threadIdx.x;
    if (v >= MPAD) return;
    unsigned short* row = A1 + (size_t)v * K1P;
    if (v >= NN) {
        for (int c = 0; c < K1P; ++c) row[c] = 0;
        return;
    }
    const float* xr = x + (size_t)v * DD;
    const float* mr = mx + (size_t)v * K1;
#pragma unroll 8
    for (int c = 0; c < DD; ++c) row[c] = f2bf(xr[c] + mr[c]);
    int hot = rmod[v];
#pragma unroll
    for (int j = 0; j < FIXEDSZ; ++j)
        row[DD + j] = f2bf(((j == hot) ? 1.0f : 0.0f) + mr[DD + j]);
#pragma unroll
    for (int c = K1; c < K1P; ++c) row[c] = 0;
}

__global__ void k_convW1(const float* __restrict__ W1, unsigned short* __restrict__ W1p) {
    int t = blockIdx.x * blockDim.x + threadIdx.x;
    if (t >= K1P * HH) return;
    int k = t / HH, c = t % HH;
    W1p[t] = (k < K1) ? f2bf(W1[(size_t)k * HH + c]) : (unsigned short)0;
}

__global__ void k_convW2(const float* __restrict__ W2, unsigned short* __restrict__ W2p) {
    int t = blockIdx.x * blockDim.x + threadIdx.x;
    if (t >= HH * OO) return;
    W2p[t] = f2bf(W2[t]);
}

// GEMM1: h[MPAD,256] = relu(A1[MPAD,160] @ W1p[160,256] + b1), bf16 WMMA, f32 acc.
// Block = 256 threads = 8 waves; wave (tw,cw) -> 16x16 tile; block tile = 64x32.
__global__ void __launch_bounds__(256) k_gemm1(const unsigned short* __restrict__ A1,
                                               const unsigned short* __restrict__ W1p,
                                               const float* __restrict__ b1,
                                               float* __restrict__ h) {
    const int wave = threadIdx.x >> 5, lane = threadIdx.x & 31;
    const int m0 = blockIdx.x * 64 + (wave & 3) * 16;
    const int n0 = blockIdx.y * 32 + (wave >> 2) * 16;
    v8f acc = {0.f, 0.f, 0.f, 0.f, 0.f, 0.f, 0.f, 0.f};
    const int arow = m0 + (lane & 15);
    const int koff = (lane < 16) ? 0 : 8;   // A fragment half-wave K offset (ISA 7.12.2)
#pragma unroll
    for (int kc = 0; kc < K1P / 32; ++kc) {
        const int k0 = kc * 32;
        Frag a, b;
        const unsigned short* ap = A1 + (size_t)arow * K1P + k0 + koff;
        a.u[0] = *(const uint4*)(ap);        // K = k0+koff + {0..7}
        a.u[1] = *(const uint4*)(ap + 16);   // K = k0+koff + {16..23}
        const unsigned short* bp = W1p + (size_t)(k0 + koff) * HH + n0 + (lane & 15);  // col n, K = k0+koff+{0..7,16..23}
        unsigned short* bs = (unsigned short*)&b;
#pragma unroll
        for (int i = 0; i < 8; ++i) { bs[i] = bp[(size_t)i * HH]; bs[8 + i] = bp[(size_t)(16 + i) * HH]; }
        acc = __builtin_amdgcn_wmma_f32_16x16x32_bf16(false, a.v, false, b.v,
                                                      (short)0, acc, false, false);
    }
    const int rbase = m0 + ((lane < 16) ? 0 : 8);
    const int col = n0 + (lane & 15);
    const float bias = b1[col];
#pragma unroll
    for (int rr = 0; rr < 8; ++rr) {
        float val = acc[rr] + bias;
        val = val > 0.f ? val : 0.f;
        h[(size_t)(rbase + rr) * HH + col] = val;
    }
}

// layer-2 messages: m2[dst] += h[src]; 64 lanes per edge, float4 per lane
__global__ void k_edge2(const int* __restrict__ src, const int* __restrict__ dst,
                        const float* __restrict__ h, float* __restrict__ m2) {
    int t = blockIdx.x * blockDim.x + threadIdx.x;
    int e = t >> 6, lane = t & 63;
    if (e >= EE) return;
    int s = src[e], d = dst[e];
    const float4 v = *(const float4*)(h + (size_t)s * HH + lane * 4);
    float* mrow = m2 + (size_t)d * HH + lane * 4;
    atomicAdd(mrow + 0, v.x);
    atomicAdd(mrow + 1, v.y);
    atomicAdd(mrow + 2, v.z);
    atomicAdd(mrow + 3, v.w);
}

// A2 = bf16(h + m2); pad rows zeroed
__global__ void k_buildA2(const float* __restrict__ h, const float* __restrict__ m2,
                          unsigned short* __restrict__ A2) {
    int v = blockIdx.x * blockDim.x + threadIdx.x;
    if (v >= MPAD) return;
    uint4* row = (uint4*)(A2 + (size_t)v * HH);
    if (v >= NN) {
        uint4 z = {0u, 0u, 0u, 0u};
#pragma unroll 8
        for (int c = 0; c < HH / 8; ++c) row[c] = z;
        return;
    }
    const float* hr = h  + (size_t)v * HH;
    const float* mr = m2 + (size_t)v * HH;
#pragma unroll 4
    for (int c = 0; c < HH / 8; ++c) {
        float4 a0 = *(const float4*)(hr + c * 8 + 0);
        float4 a1 = *(const float4*)(hr + c * 8 + 4);
        float4 b0 = *(const float4*)(mr + c * 8 + 0);
        float4 b1 = *(const float4*)(mr + c * 8 + 4);
        uint4 out;
        out.x = (uint32_t)f2bf(a0.x + b0.x) | ((uint32_t)f2bf(a0.y + b0.y) << 16);
        out.y = (uint32_t)f2bf(a0.z + b0.z) | ((uint32_t)f2bf(a0.w + b0.w) << 16);
        out.z = (uint32_t)f2bf(a1.x + b1.x) | ((uint32_t)f2bf(a1.y + b1.y) << 16);
        out.w = (uint32_t)f2bf(a1.z + b1.z) | ((uint32_t)f2bf(a1.w + b1.w) << 16);
        row[c] = out;
    }
}

// GEMM2 + fused global_add_pool: out[batch[row]] += relu(A2 @ W2p + b2)
__global__ void __launch_bounds__(256) k_gemm2(const unsigned short* __restrict__ A2,
                                               const unsigned short* __restrict__ W2p,
                                               const float* __restrict__ b2,
                                               const int* __restrict__ batch,
                                               float* __restrict__ out) {
    const int wave = threadIdx.x >> 5, lane = threadIdx.x & 31;
    const int m0 = blockIdx.x * 64 + (wave & 3) * 16;
    const int n0 = blockIdx.y * 32 + (wave >> 2) * 16;
    v8f acc = {0.f, 0.f, 0.f, 0.f, 0.f, 0.f, 0.f, 0.f};
    const int arow = m0 + (lane & 15);
    const int koff = (lane < 16) ? 0 : 8;
#pragma unroll
    for (int kc = 0; kc < HH / 32; ++kc) {
        const int k0 = kc * 32;
        Frag a, b;
        const unsigned short* ap = A2 + (size_t)arow * HH + k0 + koff;
        a.u[0] = *(const uint4*)(ap);
        a.u[1] = *(const uint4*)(ap + 16);
        const unsigned short* bp = W2p + (size_t)(k0 + koff) * OO + n0 + (lane & 15);
        unsigned short* bs = (unsigned short*)&b;
#pragma unroll
        for (int i = 0; i < 8; ++i) { bs[i] = bp[(size_t)i * OO]; bs[8 + i] = bp[(size_t)(16 + i) * OO]; }
        acc = __builtin_amdgcn_wmma_f32_16x16x32_bf16(false, a.v, false, b.v,
                                                      (short)0, acc, false, false);
    }
    const int rbase = m0 + ((lane < 16) ? 0 : 8);
    const int col = n0 + (lane & 15);
    const float bias = b2[col];
#pragma unroll
    for (int rr = 0; rr < 8; ++rr) {
        int row = rbase + rr;
        if (row < NN) {
            float val = acc[rr] + bias;
            val = val > 0.f ? val : 0.f;
            int g = batch[row];
            atomicAdd(out + (size_t)g * OO + col, val);
        }
    }
}

// ---------------- host launcher ----------------
extern "C" void kernel_launch(void* const* d_in, const int* in_sizes, int n_in,
                              void* d_out, int out_size, void* d_ws, size_t ws_size,
                              hipStream_t stream) {
    const float* x     = (const float*)d_in[0];                 // [N,128]
    const int*   src   = (const int*)d_in[1];                   // edge_index[0], E
    const int*   dst   = ((const int*)d_in[1]) + EE;            // edge_index[1], E
    const int*   batch = (const int*)d_in[2];                   // [N]
    const float* W1    = (const float*)d_in[3];                 // [138,256]
    const float* b1    = (const float*)d_in[4];                 // [256]
    const float* W2    = (const float*)d_in[5];                 // [256,256]
    const float* b2    = (const float*)d_in[6];                 // [256]
    float* out = (float*)d_out;                                 // [2000,256]

    // workspace layout (256B aligned regions)
    uint8_t* ws = (uint8_t*)d_ws;
    size_t off = 0;
    auto take = [&](size_t bytes) -> uint8_t* {
        uint8_t* p = ws + off;
        off = (off + bytes + 255) & ~(size_t)255;
        return p;
    };
    float*          mx     = (float*)take((size_t)MPAD * K1 * 4);   // 55.3 MB
    float*          rbuf   = (float*)take((size_t)MPAD * 4);
    int*            rmod   = (int*)take((size_t)MPAD * 4);
    int*            starts = (int*)take((size_t)(GG + 1) * 4);
    unsigned short* A1     = (unsigned short*)take((size_t)MPAD * K1P * 2);  // 32 MB
    unsigned short* W1p    = (unsigned short*)take((size_t)K1P * HH * 2);
    unsigned short* W2p    = (unsigned short*)take((size_t)HH * OO * 2);
    float*          h      = (float*)take((size_t)MPAD * HH * 4);   // 102 MB
    float*          m2     = (float*)take((size_t)MPAD * HH * 4);   // 102 MB
    unsigned short* A2     = (unsigned short*)take((size_t)MPAD * HH * 2);   // 51 MB
    (void)ws_size; (void)n_in; (void)in_sizes; (void)out_size;

    const int TB = 256;
    // 1) zero accumulators + output (graph-capture safe, deterministic each call)
    k_zero<<<4096, TB, 0, stream>>>(mx, (long)MPAD * K1);
    k_zero<<<4096, TB, 0, stream>>>(m2, (long)MPAD * HH);
    k_zero<<<(GG * OO + TB - 1) / TB, TB, 0, stream>>>(out, (long)GG * OO);

    // 2) segment starts via binary search over sorted batch_ids
    k_starts<<<(GG + 1 + TB - 1) / TB, TB, 0, stream>>>(batch, starts);

    // 3) JAX-compatible RNG + within-graph ranks
    k_rng<<<(NN + TB - 1) / TB, TB, 0, stream>>>(rbuf);
    k_rank<<<(NN + TB - 1) / TB, TB, 0, stream>>>(batch, starts, rbuf, rmod);

    // 4) weight conversion to padded bf16
    k_convW1<<<(K1P * HH + TB - 1) / TB, TB, 0, stream>>>(W1, W1p);
    k_convW2<<<(HH * OO + TB - 1) / TB, TB, 0, stream>>>(W2, W2p);

    // 5) layer-1 message scatter (x + one-hot counts), build A1, GEMM1
    k_edge1<<<(EE * 32 + TB - 1) / TB, TB, 0, stream>>>(src, dst, x, rmod, mx);
    k_buildA1<<<MPAD / TB, TB, 0, stream>>>(x, mx, rmod, A1);
    {
        dim3 grid(MPAD / 64, HH / 32);
        k_gemm1<<<grid, TB, 0, stream>>>(A1, W1p, b1, h);
    }

    // 6) layer-2 message scatter, build A2, GEMM2 with fused pooling
    k_edge2<<<(EE * 64 + TB - 1) / TB, TB, 0, stream>>>(src, dst, h, m2);
    k_buildA2<<<MPAD / TB, TB, 0, stream>>>(h, m2, A2);
    {
        dim3 grid(MPAD / 64, OO / 32);
        k_gemm2<<<grid, TB, 0, stream>>>(A2, W2p, b2, batch, out);
    }
}
